// PDELayer_3298534883509
// MI455X (gfx1250) — matrix operand order repaired and checked by reading, not compiled, not run on hardware
//
#include <hip/hip_runtime.h>

// ---------------------------------------------------------------------------
// PDELayer: 100 steps of variable-coefficient 2-D diffusion with frozen
// reflect halo, batch 8192 of 48x48 grids.
//
// The reference map is LINEAR in u0 and identical across the batch:
//   out[b] = K * u0[b],  K in R^{2304x2304}.
// Phase 1: build K by evolving the 2304 one-hot basis grids (exact reference
//          algorithm) fully LDS-resident; store rows of Kc in f16.
// Phase 2: Out(8192x2304) = U0(8192x2304) * Kc(2304x2304) via
//          v_wmma_f32_16x16x32_f16 (f32 accumulate), double-buffered LDS
//          software pipeline. 87 GFLOP on the matrix pipes; U0+Kc are
//          L2-resident (86 MB << 192 MB L2), HBM traffic is the compulsory
//          ~160 MB (~7 us at 23.3 TB/s).
// ---------------------------------------------------------------------------

typedef __attribute__((ext_vector_type(16))) _Float16 v16h;
typedef __attribute__((ext_vector_type(8)))  float    v8f;

namespace {
constexpr int   kN   = 48;            // grid edge
constexpr int   kNC  = kN * kN;       // 2304 cells
constexpr int   kB   = 8192;          // batch
constexpr int   kNT  = 100;           // time steps
constexpr float kDT  = 0.0001f;
constexpr float kInvDX2 = 48.0f * 48.0f;   // 1/DX^2 = 1/DY^2 = 2304
constexpr float kTwoPi  = 6.28318530717958647692f;
}

// ===========================================================================
// Phase 1: evolve one-hot basis grid n -> row n of Kc (f16)
// ===========================================================================
__global__ __launch_bounds__(256) void evolve_basis_kernel(
    const float* a1p, const float* a2p, const float* a3p,
    const float* b1p, const float* b2p, const float* b3p,
    _Float16* __restrict__ Kc)
{
  constexpr int P = 50;   // padded edge
  constexpr int S = 52;   // LDS row stride (floats)
  __shared__ float buf0[P * S];
  __shared__ float buf1[P * S];
  __shared__ float s_alpha[kN];
  __shared__ float s_beta[kN];

  const int t = threadIdx.x;
  const int n = blockIdx.x;           // basis (input cell) index

  // coefficient rows/cols (linspace(0,1,48) -> i/47)
  if (t < kN) {
    const float a1 = fabsf(*a1p), a2 = fabsf(*a2p), a3 = fabsf(*a3p);
    const float y  = (float)t * (1.0f / 47.0f);
    s_alpha[t] = 0.5f * kDT * (a1 + a2 * sinf(kTwoPi * y)
                                  + a3 * cosf(kTwoPi * y)) * kInvDX2;
  } else if (t < 2 * kN) {
    const float b1 = fabsf(*b1p), b2 = fabsf(*b2p), b3 = fabsf(*b3p);
    const int   j  = t - kN;
    const float x  = (float)j * (1.0f / 47.0f);
    s_beta[j] = kDT * (b1 + b2 * cosf(kTwoPi * x)
                          + b3 * sinf(kTwoPi * x)) * kInvDX2;
  }

  // reflect-pad of one-hot e_n: padded[i][j] = e_n[ref(i)][ref(j)]
  for (int idx = t; idx < P * P; idx += 256) {
    const int i  = idx / P, j = idx % P;
    const int si = (i == 0) ? 1 : ((i == P - 1) ? kN - 2 : i - 1);
    const int sj = (j == 0) ? 1 : ((j == P - 1) ? kN - 2 : j - 1);
    const float v = (si * kN + sj == n) ? 1.0f : 0.0f;
    buf0[i * S + j] = v;          // halo ring stays frozen in both buffers
    buf1[i * S + j] = v;
  }
  __syncthreads();

  // 2304 interior cells / 256 threads = 9 cells each; cache coords/coeffs
  int   off[9];
  float al[9], be[9];
#pragma unroll
  for (int r = 0; r < 9; ++r) {
    const int c  = t + 256 * r;
    const int ci = c / kN, cj = c % kN;
    off[r] = (ci + 1) * S + (cj + 1);
    al[r]  = s_alpha[ci];
    be[r]  = s_beta[cj];
  }

  float* cur = buf0;
  float* nxt = buf1;
  for (int step = 0; step < kNT; ++step) {
#pragma unroll
    for (int r = 0; r < 9; ++r) {
      const int   o  = off[r];
      const float ui = cur[o];
      const float sx = cur[o - S] + cur[o + S] - 2.0f * ui;
      const float sy = cur[o - 1] + cur[o + 1] - 2.0f * ui;
      nxt[o] = ui + al[r] * sx + be[r] * sy;
    }
    __syncthreads();
    float* tmp = cur; cur = nxt; nxt = tmp;
  }

  // row n of Kc (GEMM B operand: B[n][m]); coalesced f16 stores
#pragma unroll
  for (int r = 0; r < 9; ++r) {
    const int c = t + 256 * r;
    Kc[(size_t)n * kNC + c] = (_Float16)cur[off[r]];
  }
}

// ===========================================================================
// Phase 2: Out = U0 * Kc   (M=8192, N=2304, K=2304), f16 WMMA, f32 accum,
//          double-buffered LDS software pipeline.
// ===========================================================================
union FragH {
  uint4 q[2];
  v16h  h;
};

struct APref { float4 f[4]; };    // 16 f32 of one A row half
struct BPref { uint4 r0, r1; };   // 8 f16 x 2 consecutive K rows of B

__device__ inline unsigned pk_f16(float x, float y) {
  const _Float16 hx = (_Float16)x, hy = (_Float16)y;
  return (unsigned)__builtin_bit_cast(unsigned short, hx) |
         ((unsigned)__builtin_bit_cast(unsigned short, hy) << 16);
}

__global__ __launch_bounds__(256) void gemm_kernel(
    const float*    __restrict__ U0,   // [8192][2304] f32
    const _Float16* __restrict__ Kc,   // [2304][2304] f16, B[n][m]
    float*          __restrict__ Out)  // [8192][2304] f32
{
  constexpr int BM = 128, BN = 128, BK = 32;
  constexpr int KSTEPS = kNC / BK;              // 72
  constexpr int LDT = 40;                       // f16 row stride (80 B, 16B-aligned)
  __shared__ __align__(16) unsigned short sA[2][BM * LDT];  // A tiles [m][k]
  __shared__ __align__(16) unsigned short sB[2][BN * LDT];  // B tiles transposed [n][k]

  const int tid  = threadIdx.x;
  const int m0   = blockIdx.y * BM;
  const int n0   = blockIdx.x * BN;
  const int lane = tid & 31;
  const int wid  = tid >> 5;
  const int l15  = lane & 15;
  const int wm0  = (wid & 1) * 64;              // wave 64x32 patch
  const int wn0  = (wid >> 1) * 32;
  const int aoff = (lane < 16) ? 0 : 8;         // A frag K base (f16 units)
  const int bko  = (lane < 16) ? 0 : 8;         // B frag K base

  // cooperative-load index split
  const int amm   = tid >> 1;                   // A: row within tile
  const int ahalf = (tid & 1) * 16;             // A: k half (16 f32 each)
  const int bseg  = tid & 15;                   // B: 8-col segment
  const int bkk   = (tid >> 4) * 2;             // B: k row pair

  const float*    gA = U0 + (size_t)(m0 + amm) * kNC + ahalf;
  const _Float16* gB = Kc + (size_t)bkk * kNC + n0 + bseg * 8;

  v8f acc[4][2] = {};

  // ---- pipeline helpers ----------------------------------------------------
  auto load_tiles = [&](int k0, APref& pa, BPref& pb) {
    const float4* ga = (const float4*)(gA + k0);
    pa.f[0] = ga[0]; pa.f[1] = ga[1]; pa.f[2] = ga[2]; pa.f[3] = ga[3];
    pb.r0 = *(const uint4*)(gB + (size_t)k0 * kNC);
    pb.r1 = *(const uint4*)(gB + (size_t)(k0 + 1) * kNC);
  };

  auto store_tiles = [&](int stage, const APref& pa, const BPref& pb) {
    const uint4 p0 = {pk_f16(pa.f[0].x, pa.f[0].y), pk_f16(pa.f[0].z, pa.f[0].w),
                      pk_f16(pa.f[1].x, pa.f[1].y), pk_f16(pa.f[1].z, pa.f[1].w)};
    const uint4 p1 = {pk_f16(pa.f[2].x, pa.f[2].y), pk_f16(pa.f[2].z, pa.f[2].w),
                      pk_f16(pa.f[3].x, pa.f[3].y), pk_f16(pa.f[3].z, pa.f[3].w)};
    *(uint4*)&sA[stage][amm * LDT + ahalf]     = p0;
    *(uint4*)&sA[stage][amm * LDT + ahalf + 8] = p1;

    const unsigned w0[4] = {pb.r0.x, pb.r0.y, pb.r0.z, pb.r0.w};
    const unsigned w1[4] = {pb.r1.x, pb.r1.y, pb.r1.z, pb.r1.w};
#pragma unroll
    for (int e = 0; e < 8; ++e) {
      const unsigned lo = (w0[e >> 1] >> ((e & 1) * 16)) & 0xffffu;
      const unsigned hi = (w1[e >> 1] >> ((e & 1) * 16)) & 0xffffu;
      const int nn = bseg * 8 + e;
      *(unsigned*)&sB[stage][nn * LDT + bkk] = lo | (hi << 16);
    }
  };

  auto compute_tile = [&](int stage) {
    FragH a[4], b[2];
#pragma unroll
    for (int mi = 0; mi < 4; ++mi) {
      const int row = wm0 + mi * 16 + l15;
      a[mi].q[0] = *(const uint4*)&sA[stage][row * LDT + aoff];       // K aoff..+7
      a[mi].q[1] = *(const uint4*)&sA[stage][row * LDT + aoff + 16];  // K aoff+16..+23
    }
#pragma unroll
    for (int ni = 0; ni < 2; ++ni) {
      const int col = wn0 + ni * 16 + l15;
      b[ni].q[0] = *(const uint4*)&sB[stage][col * LDT + bko];        // K bko..+7
      b[ni].q[1] = *(const uint4*)&sB[stage][col * LDT + bko + 16];   // K bko+16..+23
    }
#pragma unroll
    for (int mi = 0; mi < 4; ++mi)
#pragma unroll
      for (int ni = 0; ni < 2; ++ni)
        acc[mi][ni] = __builtin_amdgcn_wmma_f32_16x16x32_f16(
            false, a[mi].h, false, b[ni].h, (short)0, acc[mi][ni],
            false, false);
  };

  // ---- prologue: tile 0 -> stage 0 ----------------------------------------
  APref pa; BPref pb;
  load_tiles(0, pa, pb);
  store_tiles(0, pa, pb);
  __syncthreads();

  // ---- pipelined main loop -------------------------------------------------
  // iter k: prefetch tile k+1 (global), compute from stage k&1 (WMMA covers
  // the prefetch latency), then convert+store prefetch into stage (k+1)&1.
  // One barrier per iteration: reads of stage s in iter k are fenced from the
  // writes to stage s in iter k+1 by the iter-k barrier.
  for (int k = 0; k < KSTEPS; ++k) {
    const bool more = (k + 1) < KSTEPS;
    if (more) load_tiles((k + 1) * BK, pa, pb);
    compute_tile(k & 1);
    if (more) store_tiles((k + 1) & 1, pa, pb);
    __syncthreads();
  }

  // ---- epilogue: C/D layout: VGPR r -> M = r + (lane<16 ? 0 : 8) -----------
  const int rsel = (lane < 16) ? 0 : 8;
#pragma unroll
  for (int mi = 0; mi < 4; ++mi)
#pragma unroll
    for (int ni = 0; ni < 2; ++ni) {
      const int col = n0 + wn0 + ni * 16 + l15;
#pragma unroll
      for (int r = 0; r < 8; ++r) {
        const int row = m0 + wm0 + mi * 16 + r + rsel;
        Out[(size_t)row * kNC + col] = acc[mi][ni][r];
      }
    }
}

// ===========================================================================
extern "C" void kernel_launch(void* const* d_in, const int* in_sizes, int n_in,
                              void* d_out, int out_size, void* d_ws, size_t ws_size,
                              hipStream_t stream) {
  (void)in_sizes; (void)n_in; (void)out_size; (void)ws_size;
  const float* u0  = (const float*)d_in[0];
  const float* aw1 = (const float*)d_in[1];
  const float* aw2 = (const float*)d_in[2];
  const float* aw3 = (const float*)d_in[3];
  const float* bw1 = (const float*)d_in[4];
  const float* bw2 = (const float*)d_in[5];
  const float* bw3 = (const float*)d_in[6];

  _Float16* Kc = (_Float16*)d_ws;              // 2304*2304*2 = 10.6 MB scratch

  evolve_basis_kernel<<<kNC, 256, 0, stream>>>(aw1, aw2, aw3, bw1, bw2, bw3, Kc);

  dim3 grid(kNC / 128, kB / 128);              // 18 x 64 blocks
  gemm_kernel<<<grid, 256, 0, stream>>>(u0, Kc, (float*)d_out);
}
